// MOLVAE_42923903156743
// MI455X (gfx1250) — matrix operand order repaired and checked by reading, not compiled, not run on hardware
//
#include <hip/hip_runtime.h>
#include <hip/hip_bf16.h>
#include <stdint.h>

// ---------------- problem constants (from reference) ----------------
#define HEADS 4
#define ENC 64
#define LAT 64
#define EDGE_DIM 11
#define DECW 128
#define N_GRAPHS 2048
#define NPG 32
#define N_NODES (N_GRAPHS * NPG)      // 65536
#define DEG 4
#define N_EDGES (N_NODES * DEG)       // 262144
#define P_PAIRS 496                   // 32*31/2
#define R_ROWS ((size_t)N_GRAPHS * P_PAIRS)  // 1015808
#define QKVS_N 832                    // 256 q + 256 k + 256 v + 64 skip

typedef __bf16 bf16;
typedef __attribute__((ext_vector_type(16))) __bf16 v16bf;
typedef __attribute__((ext_vector_type(8)))  float  v8f;

union Frag { v16bf v; uint4 q[2]; };

// ---------------- weight pre-swizzle into B-fragment layout ----------------
// B frag (this code's K order, A matches) for v_wmma_f32_16x16x32_bf16 (32x16 tile): lane L: n = L&15,
// g = L>>4 selects K half [16g,16g+16); dword v (0..7) holds K = 16g+2v, +1.
// Stored so each lane reads 8 contiguous dwords: idx = ((kt*NT + nt)*32 + lane)*8 + v
__global__ void k_swz(const float* __restrict__ W, int Ksrc, int Nsrc,
                      int Ktiles, int NTtot, int ntOff, uint32_t* __restrict__ dst)
{
    int idx = blockIdx.x * blockDim.x + threadIdx.x;
    int NTloc = Nsrc >> 4;
    int v = idx & 7;
    int lane = (idx >> 3) & 31;
    int t = idx >> 8;
    int kt = t / NTloc;
    int ntl = t - kt * NTloc;
    if (kt >= Ktiles) return;
    int g = lane >> 4, n16 = lane & 15;
    int k0 = kt * 32 + 16 * g + 2 * v;
    int col = ntl * 16 + n16;
    float lo = (k0     < Ksrc) ? W[(size_t)k0 * Nsrc + col]       : 0.f;
    float hi = (k0 + 1 < Ksrc) ? W[(size_t)(k0 + 1) * Nsrc + col] : 0.f;
    union { bf16 h[2]; uint32_t u; } pk;
    pk.h[0] = (bf16)lo; pk.h[1] = (bf16)hi;
    dst[(((size_t)kt * NTtot + (ntOff + ntl)) * 32 + lane) * 8 + v] = pk.u;
}

// ---------------- generic WMMA GEMM ----------------
// A: M x K row-major bf16, K = KTILES*32 (compile-time). Block = 4 waves; each
// wave computes a 64(M) x 64(N) C tile (4 row-frags x 4 n-tiles); block tile
// 256 x 64. Per k-tile: preload A[4] + B[4] frags (one clause), then a 16-WMMA
// burst with full A/B register reuse. k-loop fully unrolled (KTILES constant)
// so the scheduler can hoist next-tile loads under the current WMMA burst.
// flags: 1=bf16 out, 2=col stats, 4=bias.
template <int KTILES>
__global__ __launch_bounds__(128) void k_gemm(
    const bf16* __restrict__ A, const uint32_t* __restrict__ Bsw,
    const float* __restrict__ bias,
    float* __restrict__ Cf, bf16* __restrict__ Cb,
    float* __restrict__ colsum, float* __restrict__ colsumsq,
    int NT, int ldc, int flags)
{
    const int lane = threadIdx.x & 31;
    const int wave = threadIdx.x >> 5;
    const int g = lane >> 4;
    const int l16 = lane & 15;
    const long m0 = (long)blockIdx.x * 256 + wave * 64;
    const int ntBase = blockIdx.y * 4;
    const int K = KTILES * 32;
    v8f acc[4][4] = {};
    const uint32_t* arow[4];
#pragma unroll
    for (int rf = 0; rf < 4; ++rf)
        arow[rf] = (const uint32_t*)(A + (size_t)(m0 + 16 * rf + l16) * K);
#pragma unroll
    for (int kt = 0; kt < KTILES; ++kt) {
        Frag a[4], b[4];
#pragma unroll
        for (int rf = 0; rf < 4; ++rf) {
            const uint32_t* rp = arow[rf] + kt * 16;  // 32 bf16 = 16 dwords / ktile
            a[rf].q[0] = *(const uint4*)(rp + 8 * g);     // K = 16g .. 16g+7
            a[rf].q[1] = *(const uint4*)(rp + 8 * g + 4); // K = 16g+8 .. 16g+15
        }
#pragma unroll
        for (int j = 0; j < 4; ++j) {
            const uint32_t* bp = Bsw + (((size_t)kt * NT + (ntBase + j)) * 32 + lane) * 8;
            b[j].q[0] = *(const uint4*)(bp);
            b[j].q[1] = *(const uint4*)(bp + 4);
        }
#pragma unroll
        for (int j = 0; j < 4; ++j)
#pragma unroll
            for (int rf = 0; rf < 4; ++rf)
                acc[rf][j] = __builtin_amdgcn_wmma_f32_16x16x32_bf16(
                    false, a[rf].v, false, b[j].v, (short)0, acc[rf][j], false, false);
    }
#pragma unroll
    for (int j = 0; j < 4; ++j) {
        int n = (ntBase + j) * 16 + l16;
        float bv = (flags & 4) ? bias[n] : 0.f;
        float s = 0.f, s2 = 0.f;
#pragma unroll
        for (int rf = 0; rf < 4; ++rf) {
#pragma unroll
            for (int r = 0; r < 8; ++r) {
                float val = acc[rf][j][r] + bv;
                size_t o = (size_t)(m0 + 16 * rf + r + 8 * g) * ldc + n;
                if (flags & 1) Cb[o] = (bf16)val; else Cf[o] = val;
                s += val; s2 += val * val;
            }
        }
        if (flags & 2) { atomicAdd(&colsum[n], s); atomicAdd(&colsumsq[n], s2); }
    }
}

// ---------------- decoder pair-gather GEMM (K=128 fixed) ----------------
// Row r -> graph g=r/496, pair p; A row = [z[g*32+pi[p]] | z[g*32+pj[p]]] (64+64 bf16).
// Same 64x64 wave tile / 256-row block tile as k_gemm.
__global__ __launch_bounds__(128) void k_gemm_gather(
    const bf16* __restrict__ Z, const int* __restrict__ pi, const int* __restrict__ pj,
    const uint32_t* __restrict__ Bsw, const float* __restrict__ bias,
    bf16* __restrict__ Cb, float* __restrict__ colsum, float* __restrict__ colsumsq,
    int NT, int ldc)
{
    const int lane = threadIdx.x & 31;
    const int wave = threadIdx.x >> 5;
    const int g = lane >> 4;
    const int l16 = lane & 15;
    const long m0 = (long)blockIdx.x * 256 + wave * 64;
    const int ntBase = blockIdx.y * 4;
    const bf16* rowi[4];
    const bf16* rowj[4];
#pragma unroll
    for (int rf = 0; rf < 4; ++rf) {
        long m = m0 + 16 * rf + l16;
        int gr = (int)(m / P_PAIRS);
        int p  = (int)(m - (long)gr * P_PAIRS);
        rowi[rf] = Z + ((size_t)gr * NPG + pi[p]) * LAT;
        rowj[rf] = Z + ((size_t)gr * NPG + pj[p]) * LAT;
    }
    v8f acc[4][4] = {};
#pragma unroll
    for (int kt = 0; kt < 4; ++kt) {
        Frag a[4], b[4];
#pragma unroll
        for (int rf = 0; rf < 4; ++rf) {
            const bf16* row = (kt < 2) ? rowi[rf] : rowj[rf];
            const uint32_t* rp = (const uint32_t*)(row + (kt & 1) * 32);
            a[rf].q[0] = *(const uint4*)(rp + 8 * g);
            a[rf].q[1] = *(const uint4*)(rp + 8 * g + 4);
        }
#pragma unroll
        for (int j = 0; j < 4; ++j) {
            const uint32_t* bp = Bsw + (((size_t)kt * NT + (ntBase + j)) * 32 + lane) * 8;
            b[j].q[0] = *(const uint4*)(bp);
            b[j].q[1] = *(const uint4*)(bp + 4);
        }
#pragma unroll
        for (int j = 0; j < 4; ++j)
#pragma unroll
            for (int rf = 0; rf < 4; ++rf)
                acc[rf][j] = __builtin_amdgcn_wmma_f32_16x16x32_bf16(
                    false, a[rf].v, false, b[j].v, (short)0, acc[rf][j], false, false);
    }
#pragma unroll
    for (int j = 0; j < 4; ++j) {
        int n = (ntBase + j) * 16 + l16;
        float bv = bias[n];
        float s = 0.f, s2 = 0.f;
#pragma unroll
        for (int rf = 0; rf < 4; ++rf) {
#pragma unroll
            for (int r = 0; r < 8; ++r) {
                float val = acc[rf][j][r] + bv;
                Cb[(size_t)(m0 + 16 * rf + r + 8 * g) * ldc + n] = (bf16)val;
                s += val; s2 += val * val;
            }
        }
        atomicAdd(&colsum[n], s); atomicAdd(&colsumsq[n], s2);
    }
}

// ---------------- elementwise helpers ----------------
__global__ void k_f2b(const float* __restrict__ src, bf16* __restrict__ dst, size_t n) {
    size_t i = (size_t)blockIdx.x * blockDim.x + threadIdx.x;
    if (i < n) dst[i] = (bf16)src[i];
}

__global__ void k_pad_edge(const float* __restrict__ ea, bf16* __restrict__ dst) {
    size_t i = (size_t)blockIdx.x * blockDim.x + threadIdx.x;
    if (i >= (size_t)N_EDGES * 32) return;
    int c = (int)(i & 31);
    size_t e = i >> 5;
    dst[i] = (c < EDGE_DIM) ? (bf16)ea[e * EDGE_DIM + c] : (bf16)0.f;
}

__global__ void k_bias_cat(const float* __restrict__ bq, const float* __restrict__ bk,
                           const float* __restrict__ bv, const float* __restrict__ bs,
                           float* __restrict__ dst) {
    int n = blockIdx.x * blockDim.x + threadIdx.x;
    if (n >= QKVS_N) return;
    float v;
    if      (n < 256) v = bq[n];
    else if (n < 512) v = bk[n - 256];
    else if (n < 768) v = bv[n - 512];
    else              v = bs[n - 768];
    dst[n] = v;
}

// ---------------- attention: exact per-node softmax over DEG=4 contiguous edges ----
// qkvs layout per node: [q(256) | k(256) | v(256) | skip(64)], e layout [E][H][64].
__global__ void k_attn(const float* __restrict__ qkvs, const bf16* __restrict__ ebuf,
                       const int* __restrict__ srcIdx, float* __restrict__ outb)
{
    int t = blockIdx.x * blockDim.x + threadIdx.x;
    if (t >= N_NODES * HEADS) return;
    int i = t >> 2, h = t & 3;
    const float* q = qkvs + (size_t)i * QKVS_N + h * 64;
    const float* vp[DEG];
    const bf16*  ep[DEG];
    float alpha[DEG];
#pragma unroll
    for (int e = 0; e < DEG; ++e) {
        int eid = i * DEG + e;
        int s = srcIdx[eid];
        const float* kp = qkvs + (size_t)s * QKVS_N + 256 + h * 64;
        vp[e] = qkvs + (size_t)s * QKVS_N + 512 + h * 64;
        ep[e] = ebuf + (size_t)eid * (HEADS * 64) + h * 64;
        float d = 0.f;
        for (int c = 0; c < 64; ++c) d += q[c] * (kp[c] + (float)ep[e][c]);
        alpha[e] = d * 0.125f;  // 1/sqrt(64)
    }
    float mx = alpha[0];
#pragma unroll
    for (int e = 1; e < DEG; ++e) mx = fmaxf(mx, alpha[e]);
    float w[DEG], den = 0.f;
#pragma unroll
    for (int e = 0; e < DEG; ++e) { w[e] = __expf(alpha[e] - mx); den += w[e]; }
    float inv = 1.f / den;
#pragma unroll
    for (int e = 0; e < DEG; ++e) w[e] *= inv;
    for (int c = 0; c < 64; ++c) {
        float acc = 0.f;
#pragma unroll
        for (int e = 0; e < DEG; ++e) acc += w[e] * (vp[e][c] + (float)ep[e][c]);
        atomicAdd(&outb[(size_t)i * 64 + c], acc * (1.f / HEADS));  // mean over heads
    }
}

// ---------------- skip-gate combine (+ optional BN stats) ----------------
__global__ void k_combine(const float* __restrict__ qkvs, const float* __restrict__ outb,
                          const float* __restrict__ Wbeta, float* __restrict__ dst,
                          float* __restrict__ colsum, float* __restrict__ colsumsq, int doStats)
{
    int i = blockIdx.x * blockDim.x + threadIdx.x;
    if (i >= N_NODES) return;
    const float* o = outb + (size_t)i * 64;
    const float* xr = qkvs + (size_t)i * QKVS_N + 768;
    float s = 0.f;
    for (int c = 0; c < 64; ++c) {
        float oc = o[c], xc = xr[c];
        s += oc * Wbeta[c] + xc * Wbeta[64 + c] + (oc - xc) * Wbeta[128 + c];
    }
    float b = 1.f / (1.f + __expf(-s));
    for (int c = 0; c < 64; ++c) {
        float oc = o[c], xc = xr[c];
        float h = b * xc + (1.f - b) * oc;
        dst[(size_t)i * 64 + c] = h;
        if (doStats) { atomicAdd(&colsum[c], h); atomicAdd(&colsumsq[c], h * h); }
    }
}

__global__ void k_bnrelu_enc(const float* __restrict__ h, const float* __restrict__ colsum,
                             const float* __restrict__ colsumsq, const float* __restrict__ gamma,
                             const float* __restrict__ beta, bf16* __restrict__ dst)
{
    size_t i = (size_t)blockIdx.x * blockDim.x + threadIdx.x;
    if (i >= (size_t)N_NODES * 64) return;
    int c = (int)(i & 63);
    float m = colsum[c] * (1.f / N_NODES);
    float var = colsumsq[c] * (1.f / N_NODES) - m * m;
    float v = (h[i] - m) * rsqrtf(var + 1e-5f) * gamma[c] + beta[c];
    dst[i] = (bf16)fmaxf(v, 0.f);
}

__global__ void k_bnrelu_dec(bf16* __restrict__ buf, const float* __restrict__ colsum,
                             const float* __restrict__ colsumsq, const float* __restrict__ gamma,
                             const float* __restrict__ beta)
{
    size_t i = (size_t)blockIdx.x * blockDim.x + threadIdx.x;
    if (i >= R_ROWS * DECW) return;
    int c = (int)(i & (DECW - 1));
    const float invR = 1.f / (float)R_ROWS;
    float m = colsum[c] * invR;
    float var = colsumsq[c] * invR - m * m;
    float v = ((float)buf[i] - m) * rsqrtf(var + 1e-5f) * gamma[c] + beta[c];
    buf[i] = (bf16)fmaxf(v, 0.f);
}

__global__ void k_reparam(const float* __restrict__ mu, const float* __restrict__ lv,
                          const float* __restrict__ eps, bf16* __restrict__ zbf)
{
    size_t i = (size_t)blockIdx.x * blockDim.x + threadIdx.x;
    if (i >= (size_t)N_NODES * LAT) return;
    zbf[i] = (bf16)(mu[i] + eps[i] * __expf(0.5f * lv[i]));
}

__global__ void k_pairs(int* __restrict__ pi, int* __restrict__ pj) {
    int p = blockIdx.x * blockDim.x + threadIdx.x;
    if (p >= P_PAIRS) return;
    int i = 0, rem = p;
    while (rem >= (NPG - 1) - i) { rem -= (NPG - 1) - i; ++i; }
    pi[p] = i;
    pj[p] = i + 1 + rem;
}

__global__ void k_decout(const bf16* __restrict__ X, const float* __restrict__ Wo,
                         const float* __restrict__ bo, float* __restrict__ logits)
{
    size_t r = (size_t)blockIdx.x * blockDim.x + threadIdx.x;
    if (r >= R_ROWS) return;
    const bf16* row = X + r * DECW;
    float s = bo[0];
    for (int c = 0; c < DECW; ++c) s += (float)row[c] * Wo[c];
    logits[r] = s;
}

// ---------------- host orchestration ----------------
extern "C" void kernel_launch(void* const* d_in, const int* in_sizes, int n_in,
                              void* d_out, int out_size, void* d_ws, size_t ws_size,
                              hipStream_t stream)
{
    (void)in_sizes; (void)n_in; (void)out_size; (void)ws_size;
    // setup_inputs() insertion order: x, edge_attr, edge_index, batch_index, eps, params..., num_graphs
    const float* X   = (const float*)d_in[0];
    const float* EA  = (const float*)d_in[1];
    const int*   SRC = (const int*)d_in[2];        // edge_index[0] = src (first E ints)
    const float* EPS = (const float*)d_in[4];
    auto F = [&](int i) { return (const float*)d_in[i]; };
    // params order: enc[4]{Wq,bq,Wk,bk,Wv,bv,We,Wskip,bskip,Wbeta,gamma,beta},
    //               mu{10}, logvar{10}, dec_hidden[2]{W,b,gamma,beta}, dec_out{W,b}

    float* out_logits = (float*)d_out;
    float* out_mu     = out_logits + R_ROWS;
    float* out_lv     = out_mu + (size_t)N_NODES * LAT;

    char* base = (char*)d_ws;
    size_t off = 0;
    auto alloc = [&](size_t bytes) -> char* {
        off = (off + 255) & ~(size_t)255;
        char* p = base + off; off += bytes; return p;
    };
    bf16*  xbf    = (bf16*)alloc((size_t)N_NODES * 64 * 2);
    bf16*  edgebf = (bf16*)alloc((size_t)N_EDGES * 32 * 2);
    float* qkvs   = (float*)alloc((size_t)N_NODES * QKVS_N * 4);
    bf16*  ebuf   = (bf16*)alloc((size_t)N_EDGES * 256 * 2);
    float* outb   = (float*)alloc((size_t)N_NODES * 64 * 4);
    uint32_t* wswQ = (uint32_t*)alloc((size_t)2 * 52 * 32 * 8 * 4);
    uint32_t* wswE = (uint32_t*)alloc((size_t)1 * 16 * 32 * 8 * 4);
    uint32_t* wswD = (uint32_t*)alloc((size_t)4 * 8 * 32 * 8 * 4);
    float* biascat = (float*)alloc(QKVS_N * 4);
    float* stats   = (float*)alloc(256 * 4);       // [0:128)=colsum, [128:256)=colsumsq
    bf16*  zbf     = (bf16*)alloc((size_t)N_NODES * LAT * 2);
    int*   pi      = (int*)alloc(P_PAIRS * 4);
    int*   pj      = (int*)alloc(P_PAIRS * 4);
    bf16*  Y1      = (bf16*)alloc(R_ROWS * DECW * 2);
    bf16*  Y2      = (bf16*)alloc(R_ROWS * DECW * 2);

    // -------- preprocessing --------
    k_f2b<<<(N_NODES * 64 + 255) / 256, 256, 0, stream>>>(X, xbf, (size_t)N_NODES * 64);
    k_pad_edge<<<(N_EDGES * 32 + 255) / 256, 256, 0, stream>>>(EA, edgebf);

    auto run_tconv = [&](int pb, float* dstH, bool doBN, const float* gamma, const float* beta) {
        // swizzle [Wq|Wk|Wv|Wskip] into one K=64, N=832 B buffer + concat bias
        k_swz<<<(2 * 16 * 256) / 256, 256, 0, stream>>>(F(pb + 0), 64, 256, 2, 52,  0, wswQ);
        k_swz<<<(2 * 16 * 256) / 256, 256, 0, stream>>>(F(pb + 2), 64, 256, 2, 52, 16, wswQ);
        k_swz<<<(2 * 16 * 256) / 256, 256, 0, stream>>>(F(pb + 4), 64, 256, 2, 52, 32, wswQ);
        k_swz<<<(2 *  4 * 256) / 256, 256, 0, stream>>>(F(pb + 7), 64,  64, 2, 52, 48, wswQ);
        k_bias_cat<<<4, 256, 0, stream>>>(F(pb + 1), F(pb + 3), F(pb + 5), F(pb + 8), biascat);
        k_swz<<<(1 * 16 * 256) / 256, 256, 0, stream>>>(F(pb + 6), EDGE_DIM, 256, 1, 16, 0, wswE);
        // qkvs = x @ [Wq|Wk|Wv|Wskip] + bias   (M=65536, K=64, N=832)
        k_gemm<2><<<dim3(N_NODES / 256, QKVS_N / 64), 128, 0, stream>>>(
            xbf, wswQ, biascat, qkvs, nullptr, nullptr, nullptr, 52, QKVS_N, 4);
        // e = edge_attr @ We   (M=262144, K=32 padded, N=256) -> bf16
        k_gemm<1><<<dim3(N_EDGES / 256, 256 / 64), 128, 0, stream>>>(
            edgebf, wswE, nullptr, nullptr, ebuf, nullptr, nullptr, 16, 256, 1);
        (void)hipMemsetAsync(outb, 0, (size_t)N_NODES * 64 * 4, stream);
        k_attn<<<(N_NODES * HEADS) / 256, 256, 0, stream>>>(qkvs, ebuf, SRC, outb);
        if (doBN) (void)hipMemsetAsync(stats, 0, 256 * 4, stream);
        k_combine<<<N_NODES / 256, 256, 0, stream>>>(
            qkvs, outb, F(pb + 9), dstH, stats, stats + 128, doBN ? 1 : 0);
        if (doBN)
            k_bnrelu_enc<<<(N_NODES * 64) / 256, 256, 0, stream>>>(
                dstH, stats, stats + 128, gamma, beta, xbf);
    };

    // -------- encoder --------
    for (int L = 0; L < 4; ++L) {
        int pb = 5 + 12 * L;
        run_tconv(pb, outb /*in-place h*/, true, F(pb + 10), F(pb + 11));
    }
    run_tconv(53, out_mu, false, nullptr, nullptr);   // mu
    run_tconv(63, out_lv, false, nullptr, nullptr);   // logvar

    // -------- decoder --------
    k_pairs<<<2, 256, 0, stream>>>(pi, pj);
    k_reparam<<<(N_NODES * LAT) / 256, 256, 0, stream>>>(out_mu, out_lv, EPS, zbf);

    // layer 1: Y1 = relu(BN(feats @ W1 + b1)), feats gathered from z
    (void)hipMemsetAsync(stats, 0, 256 * 4, stream);
    k_swz<<<(4 * 8 * 256) / 256, 256, 0, stream>>>(F(73), 128, 128, 4, 8, 0, wswD);
    k_gemm_gather<<<dim3((unsigned)(R_ROWS / 256), 2), 128, 0, stream>>>(
        zbf, pi, pj, wswD, F(74), Y1, stats, stats + 128, 8, DECW);
    k_bnrelu_dec<<<(unsigned)((R_ROWS * DECW) / 256), 256, 0, stream>>>(
        Y1, stats, stats + 128, F(75), F(76));

    // layer 2: Y2 = relu(BN(Y1 @ W2 + b2))
    (void)hipMemsetAsync(stats, 0, 256 * 4, stream);
    k_swz<<<(4 * 8 * 256) / 256, 256, 0, stream>>>(F(77), 128, 128, 4, 8, 0, wswD);
    k_gemm<4><<<dim3((unsigned)(R_ROWS / 256), 2), 128, 0, stream>>>(
        Y1, wswD, F(78), nullptr, Y2, stats, stats + 128, 8, DECW, 1 | 2 | 4);
    k_bnrelu_dec<<<(unsigned)((R_ROWS * DECW) / 256), 256, 0, stream>>>(
        Y2, stats, stats + 128, F(79), F(80));

    // output head: logits = Y2 @ Wo + bo
    k_decout<<<(unsigned)((R_ROWS + 255) / 256), 256, 0, stream>>>(Y2, F(81), F(82), out_logits);
}
